// GraphAttnModel_33294586479124
// MI455X (gfx1250) — matrix operand order repaired and checked by reading, not compiled, not run on hardware
//
#include <hip/hip_runtime.h>

// ---------------- problem constants (match reference) ----------------
#define F_DIM 256
#define H_HEADS 4
#define D_HEAD 64
#define HD_DIM 256
#define LDS_STRIDE 264   // 256 + 8 bf16 pad: 528B row = 132 words = 4 banks skew/row

typedef __attribute__((ext_vector_type(16))) __bf16 v16bf;
typedef __attribute__((ext_vector_type(8)))  float  v8f;

// ---------------- helpers ----------------
__device__ __forceinline__ v16bf load16bf(const __bf16* p) {
    union { uint4 u[2]; v16bf v; } t;
    t.u[0] = *(const uint4*)(p);
    t.u[1] = *(const uint4*)(p + 8);
    return t.v;
}

// A fragment from LDS: elements 0..15 = K ka..ka+15
__device__ __forceinline__ v16bf ldsFrag(const __bf16* base, int row, int ka) {
    union { uint4 u[2]; v16bf v; } t;
    t.u[0] = *(const uint4*)(base + row * LDS_STRIDE + ka);
    t.u[1] = *(const uint4*)(base + row * LDS_STRIDE + ka + 8);
    return t.v;
}

__device__ __forceinline__ unsigned fenc(float f) {
    unsigned b = __float_as_uint(f);
    return (b & 0x80000000u) ? ~b : (b | 0x80000000u);
}
__device__ __forceinline__ float fdec(unsigned u) {
    return __uint_as_float((u & 0x80000000u) ? (u & 0x7FFFFFFFu) : ~u);
}

__device__ __forceinline__ float blockSum256(float v, float* sh) {
#pragma unroll
    for (int o = 16; o > 0; o >>= 1) v += __shfl_xor(v, o, 32);
    __syncthreads();                       // protect previous use of sh
    if ((threadIdx.x & 31) == 0) sh[threadIdx.x >> 5] = v;
    __syncthreads();
    return sh[0] + sh[1] + sh[2] + sh[3] + sh[4] + sh[5] + sh[6] + sh[7];
}

// ---------------- 1) weight prep: transpose + bf16 hi/lo split ----------------
__global__ void __launch_bounds__(256) prep_weights(
    const float* __restrict__ Wq, const float* __restrict__ Wk,
    const float* __restrict__ Wv, const float* __restrict__ Ws,
    const float* __restrict__ bq, const float* __restrict__ bk,
    const float* __restrict__ bv, const float* __restrict__ bs,
    __bf16* __restrict__ wtHi, __bf16* __restrict__ wtLo,
    float* __restrict__ biasPack)
{
    int i = blockIdx.x * 256 + threadIdx.x;          // 4*256*256 threads
    int mat = i >> 16;
    int rem = i & 0xFFFF;
    int kIdx = rem & 0xFF;
    int nIdx = rem >> 8;
    const float* W = (mat == 0) ? Wq : (mat == 1) ? Wk : (mat == 2) ? Wv : Ws;
    float x = W[(size_t)kIdx * HD_DIM + nIdx];       // W[k][n]
    __bf16 h = (__bf16)x;
    __bf16 l = (__bf16)(x - (float)h);
    size_t o = (size_t)mat * (HD_DIM * F_DIM) + (size_t)nIdx * F_DIM + kIdx;  // Wt[n][k]
    wtHi[o] = h;
    wtLo[o] = l;
    if (i < 4 * HD_DIM) {
        const float* b = (i < 256) ? bq : (i < 512) ? bk : (i < 768) ? bv : bs;
        biasPack[i] = b[i & 0xFF];
    }
}

// ---------------- 2) fused QKV+skip GEMM via WMMA bf16 (hi/lo split) ----------------
// grid: numMTiles blocks, 256 threads (8 waves).
// Block stages one 16x256 A strip to LDS (bf16 hi/lo) once; wave w computes
// matrix (w>>1), N-tiles [(w&1)*8, (w&1)*8+8) with 8 live accumulators.
__global__ void __launch_bounds__(256) gemm_qkvs(
    const float* __restrict__ feat,
    const __bf16* __restrict__ wtHi, const __bf16* __restrict__ wtLo,
    const float* __restrict__ biasPack,
    float* __restrict__ outBase, int n)
{
    __shared__ __bf16 ldsHi[16 * LDS_STRIDE];
    __shared__ __bf16 ldsLo[16 * LDS_STRIDE];

    const int mtile = blockIdx.x;

    // ---- cooperative stage: 16 rows x 256 cols fp32 -> bf16 hi/lo in LDS ----
    {
        int t = threadIdx.x;
        int row = t >> 4;
        int colBase = (t & 15) << 4;
        int arow = mtile * 16 + row;
        if (arow >= n) arow = n - 1;                 // guard (N divisible by 16 anyway)
        const float* ap = feat + (size_t)arow * F_DIM + colBase;
        float4 f0 = *(const float4*)(ap + 0);
        float4 f1 = *(const float4*)(ap + 4);
        float4 f2 = *(const float4*)(ap + 8);
        float4 f3 = *(const float4*)(ap + 12);
        float af[16] = { f0.x,f0.y,f0.z,f0.w, f1.x,f1.y,f1.z,f1.w,
                         f2.x,f2.y,f2.z,f2.w, f3.x,f3.y,f3.z,f3.w };
        union { uint4 u[2]; __bf16 e[16]; } hi, lo;
#pragma unroll
        for (int i2 = 0; i2 < 16; ++i2) {
            float fx = af[i2];
            __bf16 h = (__bf16)fx;
            hi.e[i2] = h;
            lo.e[i2] = (__bf16)(fx - (float)h);
        }
        *(uint4*)&ldsHi[row * LDS_STRIDE + colBase + 0] = hi.u[0];
        *(uint4*)&ldsHi[row * LDS_STRIDE + colBase + 8] = hi.u[1];
        *(uint4*)&ldsLo[row * LDS_STRIDE + colBase + 0] = lo.u[0];
        *(uint4*)&ldsLo[row * LDS_STRIDE + colBase + 8] = lo.u[1];
    }
    __syncthreads();

    const int wave  = threadIdx.x >> 5;
    const int lane  = threadIdx.x & 31;
    const int row16 = lane & 15;
    const int hiK   = lane >> 4;
    const int mat   = wave >> 1;                     // 0..3
    const int ntb   = (wave & 1) * 8;                // N-tile base: 0 or 8

    const __bf16* wHiBase = wtHi + (size_t)mat * (HD_DIM * F_DIM);
    const __bf16* wLoBase = wtLo + (size_t)mat * (HD_DIM * F_DIM);
    float* out = outBase + (size_t)mat * n * HD_DIM;

    v8f zero = {};
    v8f acc[8];
#pragma unroll
    for (int i = 0; i < 8; ++i) acc[i] = zero;

#pragma unroll
    for (int kk = 0; kk < F_DIM; kk += 32) {
        const int ka = kk + hiK * 16;                // A frag: 16 sequential K per half-wave
        v16bf ahi = ldsFrag(ldsHi, row16, ka);
        v16bf alo = ldsFrag(ldsLo, row16, ka);
        const int kb = kk + hiK * 16;                // B frag: 16 sequential K per half-wave
#pragma unroll
        for (int i = 0; i < 8; ++i) {
            const int bcol = (ntb + i) * 16 + row16; // B lane column
            v16bf bh = load16bf(wHiBase + (size_t)bcol * F_DIM + kb);
            v16bf bl = load16bf(wLoBase + (size_t)bcol * F_DIM + kb);
            acc[i] = __builtin_amdgcn_wmma_f32_16x16x32_bf16(false, ahi, false, bh, (short)0, acc[i], false, false);
            acc[i] = __builtin_amdgcn_wmma_f32_16x16x32_bf16(false, ahi, false, bl, (short)0, acc[i], false, false);
            acc[i] = __builtin_amdgcn_wmma_f32_16x16x32_bf16(false, alo, false, bh, (short)0, acc[i], false, false);
        }
    }

    // C layout: VGPR r -> M = (lane>>4)*8 + r, N = lane&15
    const int mbase = mtile * 16 + hiK * 8;
#pragma unroll
    for (int i = 0; i < 8; ++i) {
        const int col = (ntb + i) * 16 + row16;
        const float b = biasPack[mat * HD_DIM + col];
#pragma unroll
        for (int r = 0; r < 8; ++r) {
            int mrow = mbase + r;
            if (mrow < n) out[(size_t)mrow * HD_DIM + col] = acc[i][r] + b;
        }
    }
}

// ---------------- 3) init: zero d_out (agg), denom, maxu ----------------
__global__ void __launch_bounds__(256) init_bufs(
    float* __restrict__ dout, float* __restrict__ denom,
    unsigned* __restrict__ maxu, int nHD, int nH)
{
    int i = blockIdx.x * 256 + threadIdx.x;
    if (i < nHD) dout[i] = 0.0f;
    if (i < nH) { denom[i] = 0.0f; maxu[i] = 0u; }
}

// ---------------- 4) per-edge QK score + segment max (wave per edge) ----------------
__global__ void __launch_bounds__(256) score_max(
    const float* __restrict__ q, const float* __restrict__ k,
    const int* __restrict__ src, const int* __restrict__ dst,
    float* __restrict__ score, unsigned* __restrict__ maxu, int E)
{
    const int wave = threadIdx.x >> 5, lane = threadIdx.x & 31;
    const int e = blockIdx.x * 8 + wave;
    if (e >= E) return;
    const int s = src[e], d = dst[e];
    const float* qp = q + (size_t)s * HD_DIM + lane * 8;
    const float* kp = k + (size_t)d * HD_DIM + lane * 8;
    float4 q0 = *(const float4*)qp, q1 = *(const float4*)(qp + 4);
    float4 k0 = *(const float4*)kp, k1 = *(const float4*)(kp + 4);
    float p = q0.x*k0.x + q0.y*k0.y + q0.z*k0.z + q0.w*k0.w
            + q1.x*k1.x + q1.y*k1.y + q1.z*k1.z + q1.w*k1.w;
    p += __shfl_xor(p, 4, 32);
    p += __shfl_xor(p, 2, 32);
    p += __shfl_xor(p, 1, 32);
    if ((lane & 7) == 0) {
        const int h = lane >> 3;
        float sc = p * 0.125f;                     // 1/sqrt(64)
        score[(size_t)e * H_HEADS + h] = sc;
        atomicMax(&maxu[(size_t)d * H_HEADS + h], fenc(sc));
    }
}

// ---------------- 5) exp(score - max), accumulate denom ----------------
__global__ void __launch_bounds__(256) exp_denom(
    const int* __restrict__ dst, float* __restrict__ score,
    const unsigned* __restrict__ maxu, float* __restrict__ denom, int EH)
{
    int i = blockIdx.x * 256 + threadIdx.x;
    if (i >= EH) return;
    int e = i >> 2, h = i & 3;
    int d = dst[e];
    float mx = fdec(maxu[(size_t)d * H_HEADS + h]);
    float a = __expf(score[i] - mx);
    score[i] = a;
    atomicAdd(&denom[(size_t)d * H_HEADS + h], a);
}

// ---------------- 6) agg[dst] += a * v[src] (wave per edge, into d_out) ----------------
__global__ void __launch_bounds__(256) aggregate(
    const float* __restrict__ v, const int* __restrict__ src,
    const int* __restrict__ dst, const float* __restrict__ score,
    float* __restrict__ dout, int E)
{
    const int wave = threadIdx.x >> 5, lane = threadIdx.x & 31;
    const int e = blockIdx.x * 8 + wave;
    if (e >= E) return;
    const int s = src[e], d = dst[e];
    const float a = score[(size_t)e * H_HEADS + (lane >> 3)];
    const float* vp = v + (size_t)s * HD_DIM + lane * 8;
    float* op = dout + (size_t)d * HD_DIM + lane * 8;
    float4 v0 = *(const float4*)vp, v1 = *(const float4*)(vp + 4);
    atomicAdd(op + 0, v0.x * a);
    atomicAdd(op + 1, v0.y * a);
    atomicAdd(op + 2, v0.z * a);
    atomicAdd(op + 3, v0.w * a);
    atomicAdd(op + 4, v1.x * a);
    atomicAdd(op + 5, v1.y * a);
    atomicAdd(op + 6, v1.z * a);
    atomicAdd(op + 7, v1.w * a);
}

// ---------------- 7) gate + LayerNorm + PReLU (block per node) ----------------
__global__ void __launch_bounds__(256) finalize(
    float* __restrict__ dout, const float* __restrict__ skip,
    const float* __restrict__ denom,
    const float* __restrict__ Wg, const float* __restrict__ bg,
    const float* __restrict__ lns, const float* __restrict__ lnb,
    const float* __restrict__ alpha)
{
    __shared__ float sred[8];
    const int nId = blockIdx.x;
    const int j = threadIdx.x;

    float agg = dout[(size_t)nId * HD_DIM + j];
    float den = denom[(size_t)nId * H_HEADS + (j >> 6)];
    float rst = (den > 0.0f) ? (agg / den) : 0.0f;
    float sk = skip[(size_t)nId * HD_DIM + j];

    float gp = sk * Wg[j] + rst * Wg[HD_DIM + j] + (sk - rst) * Wg[2 * HD_DIM + j];
    float gsum = blockSum256(gp, sred);
    float g = 1.0f / (1.0f + __expf(-(gsum + bg[0])));

    float y = g * sk + (1.0f - g) * rst;
    float s1 = blockSum256(y, sred);
    float s2 = blockSum256(y * y, sred);
    float mean = s1 * (1.0f / HD_DIM);
    float var = s2 * (1.0f / HD_DIM) - mean * mean;
    float o = (y - mean) * rsqrtf(var + 1e-5f) * lns[j] + lnb[j];
    float al = alpha[0];
    dout[(size_t)nId * HD_DIM + j] = (o > 0.0f) ? o : al * o;
}

// ---------------- launch ----------------
extern "C" void kernel_launch(void* const* d_in, const int* in_sizes, int n_in,
                              void* d_out, int out_size, void* d_ws, size_t ws_size,
                              hipStream_t stream)
{
    const float* feat = (const float*)d_in[0];
    const int*   src  = (const int*)d_in[1];
    const int*   dst  = (const int*)d_in[2];
    const float* Wq   = (const float*)d_in[3];
    const float* bq   = (const float*)d_in[4];
    const float* Wk   = (const float*)d_in[5];
    const float* bk   = (const float*)d_in[6];
    const float* Wv   = (const float*)d_in[7];
    const float* bv   = (const float*)d_in[8];
    const float* Ws   = (const float*)d_in[9];
    const float* bs   = (const float*)d_in[10];
    const float* Wg   = (const float*)d_in[11];
    const float* bg   = (const float*)d_in[12];
    const float* lns  = (const float*)d_in[13];
    const float* lnb  = (const float*)d_in[14];
    const float* alp  = (const float*)d_in[15];

    const int n = in_sizes[0] / F_DIM;     // 50000
    const int E = in_sizes[1];             // 800000

    // workspace layout (all offsets 256B aligned)
    char* ws = (char*)d_ws;
    size_t off = 0;
    __bf16* wtHi = (__bf16*)(ws + off); off += (size_t)4 * HD_DIM * F_DIM * 2;       // 512 KB
    __bf16* wtLo = (__bf16*)(ws + off); off += (size_t)4 * HD_DIM * F_DIM * 2;       // 512 KB
    float* biasPack = (float*)(ws + off); off += (size_t)4 * HD_DIM * 4;             // 4 KB
    float* qkvs = (float*)(ws + off); off += (size_t)4 * n * HD_DIM * 4;             // 4 x 51.2 MB
    float* scoreBuf = (float*)(ws + off); off += (size_t)E * H_HEADS * 4;            // 12.8 MB
    float* denom = (float*)(ws + off); off += (size_t)n * H_HEADS * 4;               // 0.8 MB
    unsigned* maxu = (unsigned*)(ws + off); off += (size_t)n * H_HEADS * 4;          // 0.8 MB

    float* q    = qkvs + (size_t)0 * n * HD_DIM;
    float* k    = qkvs + (size_t)1 * n * HD_DIM;
    float* v    = qkvs + (size_t)2 * n * HD_DIM;
    float* skip = qkvs + (size_t)3 * n * HD_DIM;
    float* out  = (float*)d_out;

    // 1) weight transpose + bf16 split
    prep_weights<<<(4 * HD_DIM * F_DIM + 255) / 256, 256, 0, stream>>>(
        Wq, Wk, Wv, Ws, bq, bk, bv, bs, wtHi, wtLo, biasPack);

    // 2) fused QKV+skip WMMA GEMM (LDS-staged A, 8 accumulators per wave)
    gemm_qkvs<<<(n + 15) / 16, 256, 0, stream>>>(feat, wtHi, wtLo, biasPack, qkvs, n);

    // 3) zero aggregation target (d_out) and per-node stats
    init_bufs<<<(n * HD_DIM + 255) / 256, 256, 0, stream>>>(
        out, denom, maxu, n * HD_DIM, n * H_HEADS);

    // 4) per-edge scores + running max
    score_max<<<(E + 7) / 8, 256, 0, stream>>>(q, k, src, dst, scoreBuf, maxu, E);

    // 5) exp + denominator
    exp_denom<<<(E * H_HEADS + 255) / 256, 256, 0, stream>>>(
        dst, scoreBuf, maxu, denom, E * H_HEADS);

    // 6) weighted aggregation into d_out
    aggregate<<<(E + 7) / 8, 256, 0, stream>>>(v, src, dst, scoreBuf, out, E);

    // 7) gate + LayerNorm + PReLU, in place on d_out
    finalize<<<n, 256, 0, stream>>>(out, skip, denom, Wg, bg, lns, lnb, alp);
}
